// MoCo_14688788152962
// MI455X (gfx1250) — matrix operand order repaired and checked by reading, not compiled, not run on hardware
//
#include <hip/hip_runtime.h>
#include <hip/hip_bf16.h>
#include <math.h>
#include <stdint.h>

typedef __attribute__((ext_vector_type(2)))  float  v2f;
typedef __attribute__((ext_vector_type(8)))  float  v8f;
typedef __attribute__((ext_vector_type(8)))  __bf16 v8bf;
typedef __attribute__((ext_vector_type(16))) __bf16 v16bf;
typedef __attribute__((ext_vector_type(4)))  unsigned int v4u;
typedef __attribute__((ext_vector_type(8)))  int    v8i_;
typedef __attribute__((ext_vector_type(4)))  int    v4i_;

#define NC     5
#define DIM    128
#define DMLP   2048
#define KQ     65536
#define ROWS   1280            // b*nc
#define TOPK   5
#define LCOLS  65537           // 1 + KQ
#define INVT   (1.0f/0.07f)
#define NCB1   512             // GEMM1 column blocks (65536/128)
#define G2_NSB 128             // GEMM2 superblocks (65536/512)
#define NEGINF (-3.0e38f)

static __device__ __forceinline__ __bf16 f2bf(float f) {
    union { float f; unsigned u; } c; c.f = f;
    unsigned r = (c.u + 0x7FFFu + ((c.u >> 16) & 1u)) >> 16;
    union { unsigned short s; __bf16 b; } o; o.s = (unsigned short)r;
    return o.b;
}

// Tensor Data Mover: 2D f32 tile (tile_x cols contiguous, tile_y rows, row stride
// stride_x elements) from global -> LDS (rows packed contiguously at lds_off).
// D# layout per CDNA5 ISA ch.8 (group0 128b, group1 256b; groups 2/3 disabled).
// This toolchain exposes the 6-arg builtin: (g0, g1, g2, g3, g4, cpol).
static __device__ __forceinline__ void tdm_load_2d_f32(
    unsigned lds_off, const void* gptr,
    unsigned tile_x, unsigned tile_y, unsigned stride_x) {
    unsigned long long ga = (unsigned long long)(uintptr_t)gptr;
    v4u g0;
    g0.x = 1u;                                               // count=1, user mode
    g0.y = lds_off;                                          // lds_addr (bytes)
    g0.z = (unsigned)(ga & 0xFFFFFFFFu);                     // global_addr[31:0]
    g0.w = (unsigned)((ga >> 32) & 0x01FFFFFFu) | (2u << 30);// addr[56:32] | type=2
    v8i_ g1;
    g1[0] = (int)(2u << 16);            // workgroup_mask=0, data_size=2 (4B)
    g1[1] = 0;                          // atomic_barrier=0, tensor_dim0[15:0]=0
    g1[2] = (int)0x00000010;            // tensor_dim0=1<<20 (hi16), tensor_dim1 lo16=0
    g1[3] = (int)(0x00000010u | (tile_x << 16)); // tensor_dim1=1<<20 (hi16), tile_dim0
    g1[4] = (int)tile_y;                // tile_dim1, tile_dim2=0
    g1[5] = (int)stride_x;              // tensor_dim0_stride[31:0]
    g1[6] = 0;                          // stride0 hi16, stride1 lo16
    g1[7] = 0;
    v4i_ z4 = {0, 0, 0, 0};
    v8i_ z8 = {0, 0, 0, 0, 0, 0, 0, 0};
    __builtin_amdgcn_tensor_load_to_lds(g0, g1, z4, z4, z8, 0);
}

// ---------------- Kernel 1: normalize q rows, compute l_pos/T into logits col 0 ----
__global__ void norm_qk_kernel(const float* __restrict__ q, const float* __restrict__ k,
                               float* __restrict__ qn, float* __restrict__ out) {
    int wid  = (blockIdx.x * blockDim.x + threadIdx.x) >> 5;   // one wave32 per row
    int lane = threadIdx.x & 31;
    if (wid >= ROWS) return;
    const float* qr = q + (size_t)wid * DIM;
    const float* kr = k + (size_t)(wid / NC) * DIM;
    float qv[4], kv[4], qss = 0.f, kss = 0.f, dot = 0.f;
#pragma unroll
    for (int j = 0; j < 4; ++j) {
        int c = lane + j * 32;
        qv[j] = qr[c]; kv[j] = kr[c];
        qss += qv[j] * qv[j]; kss += kv[j] * kv[j]; dot += qv[j] * kv[j];
    }
#pragma unroll
    for (int m = 16; m >= 1; m >>= 1) {
        qss += __shfl_xor(qss, m, 32);
        kss += __shfl_xor(kss, m, 32);
        dot += __shfl_xor(dot, m, 32);
    }
    float qd = fmaxf(sqrtf(qss), 1e-12f);
    float kd = fmaxf(sqrtf(kss), 1e-12f);
    float iq = 1.0f / qd;
#pragma unroll
    for (int j = 0; j < 4; ++j) qn[(size_t)wid * DIM + lane + j * 32] = qv[j] * iq;
    if (lane == 0) out[(size_t)wid * LCOLS] = (dot / (qd * kd)) * INVT;
}

// ---------------- Kernel 2: normalize q_backbone rows -> bf16 ----------------------
__global__ void norm_qb_kernel(const float* __restrict__ src, __bf16* __restrict__ qb) {
    __shared__ float red[256];
    int r = blockIdx.x, t = threadIdx.x;
    const float* row = src + (size_t)r * DMLP;
    float v[8], ss = 0.f;
#pragma unroll
    for (int j = 0; j < 8; ++j) { v[j] = row[t + j * 256]; ss += v[j] * v[j]; }
    red[t] = ss; __syncthreads();
    for (int s = 128; s > 0; s >>= 1) { if (t < s) red[t] += red[t + s]; __syncthreads(); }
    float inv = 1.0f / fmaxf(sqrtf(red[0]), 1e-12f);
#pragma unroll
    for (int j = 0; j < 8; ++j) qb[(size_t)r * DMLP + t + j * 256] = f2bf(v[j] * inv);
}

// ---------------- Kernel 3: GEMM1 (f32 WMMA) l_neg -> logits + softmax partials ----
// grid (512 colblocks, 10 rowblocks), 256 threads (8 waves). Wave w owns Mtile w,
// Ntiles 0..7 (128x128 tile per block). B staged by one TDM tensor load (64KB).
__global__ __launch_bounds__(256)
void gemm_qneg_kernel(const float* __restrict__ qn, const float* __restrict__ queue,
                      float* __restrict__ out, float* __restrict__ smm,
                      float* __restrict__ sms) {
    __shared__ float Bqs[128][128];            // [k][col], 64KB, TDM destination
    const int cb = blockIdx.x, rb = blockIdx.y, tid = threadIdx.x;
    const int w = tid >> 5, lane = tid & 31;
    const int n = lane & 15, h = lane >> 4;
    const int colbase = cb * 128, rowbase = rb * 128;

    if (w == 0) {
        tdm_load_2d_f32((unsigned)(uintptr_t)&Bqs[0][0], queue + colbase,
                        128u, 128u, (unsigned)KQ);
        __builtin_amdgcn_s_wait_tensorcnt(0);
    }
    __syncthreads();

    const v8f vz = {0.f,0.f,0.f,0.f,0.f,0.f,0.f,0.f};
    v8f acc[8];
#pragma unroll
    for (int t = 0; t < 8; ++t) acc[t] = vz;

    const float* ap = qn + (size_t)(rowbase + w * 16 + n) * DIM;  // n == A row (lane&15)
    for (int kb = 0; kb < DIM; kb += 4) {
        v2f a = *(const v2f*)(ap + kb + h * 2);                   // A 16x4 layout
#pragma unroll
        for (int nt = 0; nt < 8; ++nt) {
            int col = nt * 16 + n;
            v2f b; b.x = Bqs[kb + h * 2][col]; b.y = Bqs[kb + h * 2 + 1][col];
            acc[nt] = __builtin_amdgcn_wmma_f32_16x16x4_f32(
                false, a, false, b, (short)0, acc[nt], false, false);
        }
    }
    // scale by 1/T, store logits[:, 1+col]
#pragma unroll
    for (int nt = 0; nt < 8; ++nt)
#pragma unroll
        for (int v = 0; v < 8; ++v) {
            float x = acc[nt][v] * INVT;
            acc[nt][v] = x;
            int row = rowbase + w * 16 + v + 8 * h;
            int col = colbase + nt * 16 + n;
            out[(size_t)row * LCOLS + 1 + col] = x;
        }
    // per-row (max, sumexp) over this 128-col slice; reductions stay in 16-lane half
#pragma unroll
    for (int v = 0; v < 8; ++v) {
        float M = NEGINF;
#pragma unroll
        for (int nt = 0; nt < 8; ++nt) M = fmaxf(M, acc[nt][v]);
#pragma unroll
        for (int msk = 8; msk >= 1; msk >>= 1) M = fmaxf(M, __shfl_xor(M, msk, 32));
        float S = 0.f;
#pragma unroll
        for (int nt = 0; nt < 8; ++nt) S += __expf(acc[nt][v] - M);
#pragma unroll
        for (int msk = 8; msk >= 1; msk >>= 1) S += __shfl_xor(S, msk, 32);
        if (n == 0) {
            int row = rowbase + w * 16 + v + 8 * h;
            smm[(size_t)row * NCB1 + cb] = M;
            sms[(size_t)row * NCB1 + cb] = S;
        }
    }
}

// ---------------- Kernel 4: GEMM2 (bf16 WMMA) + TDM double buffer + fused top-5 ----
// grid (128 superblocks of 512 cols, 20 M-blocks of 64 rows), 256 threads (8 waves).
// Wave w: Mtile = w&3, Ntile group = w>>2. TDM loads f32 k-slab i+1 while slab i is
// converted to bf16 and consumed by WMMA. C dump aliases the staging buffers.
__global__ __launch_bounds__(256)
void gemm_qb_topk_kernel(const __bf16* __restrict__ qb, const float* __restrict__ qbk,
                         float* __restrict__ topv, int* __restrict__ topi) {
    __shared__ __align__(16) char smem[49152];     // stage(48KB) aliased with Cs(33KB)
    float  (*BsF32)[32][128] = (float  (*)[32][128])smem;           // 2 x 16KB
    __bf16 (*Bs)[128][32]    = (__bf16 (*)[128][32])(smem + 32768); // 2 x 8KB
    float  (*Cs)[129]        = (float  (*)[129])smem;               // 64x129 f32
    __shared__ float  ltv[64][TOPK];
    __shared__ int    lti[64][TOPK];

    const int sb = blockIdx.x, mb = blockIdx.y, tid = threadIdx.x;
    const int w = tid >> 5, lane = tid & 31;
    const int mt = w & 3, ntg = w >> 2;
    const int n = lane & 15, h = lane >> 4;
    const int rowbase = mb * 64;

    if (tid < 64) {
#pragma unroll
        for (int j = 0; j < TOPK; ++j) { ltv[tid][j] = NEGINF; lti[tid][j] = 0; }
    }
    __syncthreads();

    for (int chunk = 0; chunk < 4; ++chunk) {
        const int colbase = sb * 512 + chunk * 128;
        const v8f vz = {0.f,0.f,0.f,0.f,0.f,0.f,0.f,0.f};
        v8f acc[4];
#pragma unroll
        for (int t = 0; t < 4; ++t) acc[t] = vz;

        if (w == 0)                                   // prologue: slab 0
            tdm_load_2d_f32((unsigned)(uintptr_t)&BsF32[0][0][0],
                            qbk + colbase, 128u, 32u, (unsigned)KQ);

        for (int it = 0; it < DMLP / 32; ++it) {
            const int kb = it * 32, buf = it & 1;
            if (w == 0) {
                if (it + 1 < DMLP / 32) {             // issue slab i+1, wait slab i
                    tdm_load_2d_f32((unsigned)(uintptr_t)&BsF32[buf ^ 1][0][0],
                                    qbk + (size_t)(kb + 32) * KQ + colbase,
                                    128u, 32u, (unsigned)KQ);
                    __builtin_amdgcn_s_wait_tensorcnt(1);
                } else {
                    __builtin_amdgcn_s_wait_tensorcnt(0);
                }
            }
            __syncthreads();
#pragma unroll
            for (int i = 0; i < 16; ++i) {            // f32 -> bf16, transpose in LDS
                int e = i * 256 + tid;
                int kr = e >> 7, c = e & 127;
                Bs[buf][c][kr] = f2bf(BsF32[buf][kr][c]);
            }
            __syncthreads();
            const __bf16* arow = qb + (size_t)(rowbase + mt * 16 + n) * DMLP + kb + h * 16;
            v8bf alo = *(const v8bf*)arow;
            v8bf ahi = *(const v8bf*)(arow + 8);
            v16bf a = __builtin_shufflevector(alo, ahi,
                        0,1,2,3,4,5,6,7,8,9,10,11,12,13,14,15);
#pragma unroll
            for (int t = 0; t < 4; ++t) {
                int col = (ntg * 4 + t) * 16 + n;
                v16bf bm = *(const v16bf*)&Bs[buf][col][h * 16];  // B 32x16: K = h*16+e
                acc[t] = __builtin_amdgcn_wmma_f32_16x16x32_bf16(
                    false, a, false, bm, (short)0, acc[t], false, false);
            }
        }
        __syncthreads();                              // stage buffers dead; reuse as Cs
#pragma unroll
        for (int t = 0; t < 4; ++t)
#pragma unroll
            for (int v = 0; v < 8; ++v)
                Cs[mt * 16 + v + 8 * h][(ntg * 4 + t) * 16 + n] = acc[t][v];
        __syncthreads();
        if (tid < 64) {                               // per-row running top-5
            float tv[TOPK]; int ti[TOPK];
#pragma unroll
            for (int j = 0; j < TOPK; ++j) { tv[j] = ltv[tid][j]; ti[j] = lti[tid][j]; }
            for (int c = 0; c < 128; ++c) {
                float x = Cs[tid][c];
                if (x > tv[TOPK - 1]) {
                    tv[TOPK - 1] = x; ti[TOPK - 1] = colbase + c;
#pragma unroll
                    for (int j = TOPK - 1; j > 0; --j)
                        if (tv[j] > tv[j - 1]) {
                            float fv = tv[j]; tv[j] = tv[j - 1]; tv[j - 1] = fv;
                            int iv = ti[j]; ti[j] = ti[j - 1]; ti[j - 1] = iv;
                        }
                }
            }
#pragma unroll
            for (int j = 0; j < TOPK; ++j) { ltv[tid][j] = tv[j]; lti[tid][j] = ti[j]; }
        }
        __syncthreads();                              // protects Cs region before next TDM
    }
    if (tid < 64) {
        size_t base = ((size_t)(rowbase + tid) * G2_NSB + sb) * TOPK;
#pragma unroll
        for (int j = 0; j < TOPK; ++j) { topv[base + j] = ltv[tid][j]; topi[base + j] = lti[tid][j]; }
    }
}

// ---------------- Kernel 5: merge per-superblock top-5 -> knn indices --------------
__global__ void topk_merge_kernel(const float* __restrict__ topv, const int* __restrict__ topi,
                                  int* __restrict__ knn) {
    int r = blockIdx.x * blockDim.x + threadIdx.x;
    if (r >= ROWS) return;
    float tv[TOPK]; int ti[TOPK];
#pragma unroll
    for (int j = 0; j < TOPK; ++j) { tv[j] = NEGINF; ti[j] = 0; }
    for (int s = 0; s < G2_NSB; ++s) {
        size_t base = ((size_t)r * G2_NSB + s) * TOPK;
        for (int c = 0; c < TOPK; ++c) {
            float x = topv[base + c]; int idx = topi[base + c];
            if (x > tv[TOPK - 1]) {
                tv[TOPK - 1] = x; ti[TOPK - 1] = idx;
#pragma unroll
                for (int j = TOPK - 1; j > 0; --j)
                    if (tv[j] > tv[j - 1]) {
                        float fv = tv[j]; tv[j] = tv[j - 1]; tv[j - 1] = fv;
                        int iv = ti[j]; ti[j] = ti[j - 1]; ti[j - 1] = iv;
                    }
            }
        }
    }
#pragma unroll
    for (int j = 0; j < TOPK; ++j) knn[r * TOPK + j] = ti[j];
}

// ---------------- Kernel 6: per-row logsumexp merge + gather top-5 log-probs -------
__global__ void row_lse_loss_kernel(const float* __restrict__ smm, const float* __restrict__ sms,
                                    const int* __restrict__ knn, const float* __restrict__ out,
                                    float* __restrict__ row_loss) {
    int r = blockIdx.x * blockDim.x + threadIdx.x;
    if (r >= ROWS) return;
    float M = NEGINF, S = 0.f;
    for (int cb = 0; cb < NCB1; ++cb) {
        float m = smm[(size_t)r * NCB1 + cb], s = sms[(size_t)r * NCB1 + cb];
        if (m > M) { S = S * __expf(M - m) + s; M = m; }
        else       { S += s * __expf(m - M); }
    }
    float lse = M + __logf(S);
    const float* lrow = out + (size_t)r * LCOLS + 1;   // already scaled by 1/T
    float a = 0.f;
#pragma unroll
    for (int j = 0; j < TOPK; ++j) a += lrow[knn[r * TOPK + j]] - lse;
    row_loss[r] = a;
}

// ---------------- Kernel 7: labels + deterministic loss reduction ------------------
__global__ void finalize_kernel(const float* __restrict__ row_loss, float* __restrict__ out) {
    __shared__ float red[256];
    int t = threadIdx.x;
    const size_t L0 = (size_t)ROWS * LCOLS;
    for (int i = t; i < ROWS; i += 256) out[L0 + i] = 0.0f;   // int32 zeros bit-identical
    float s = 0.f;
    for (int i = t; i < ROWS; i += 256) s += row_loss[i];
    red[t] = s; __syncthreads();
    for (int st = 128; st > 0; st >>= 1) { if (t < st) red[t] += red[t + st]; __syncthreads(); }
    if (t == 0) out[L0 + ROWS] = -red[0] / (float)(ROWS * TOPK);
}

extern "C" void kernel_launch(void* const* d_in, const int* in_sizes, int n_in,
                              void* d_out, int out_size, void* d_ws, size_t ws_size,
                              hipStream_t stream) {
    const float* q       = (const float*)d_in[0];
    const float* k       = (const float*)d_in[1];
    const float* qbb     = (const float*)d_in[2];
    const float* queue   = (const float*)d_in[3];
    const float* queue_b = (const float*)d_in[4];
    float* out = (float*)d_out;

    char* ws = (char*)d_ws;
    size_t off = 0;
    auto carve = [&](size_t bytes) -> void* {
        void* p = ws + off;
        off = (off + bytes + 255) & ~(size_t)255;
        return p;
    };
    float*  qn       = (float*) carve((size_t)ROWS * DIM * 4);
    __bf16* qb       = (__bf16*)carve((size_t)ROWS * DMLP * 2);
    float*  topv     = (float*) carve((size_t)ROWS * G2_NSB * TOPK * 4);
    int*    topi     = (int*)   carve((size_t)ROWS * G2_NSB * TOPK * 4);
    int*    knn      = (int*)   carve((size_t)ROWS * TOPK * 4);
    float*  smm      = (float*) carve((size_t)ROWS * NCB1 * 4);
    float*  sms      = (float*) carve((size_t)ROWS * NCB1 * 4);
    float*  row_loss = (float*) carve((size_t)ROWS * 4);

    norm_qk_kernel<<<dim3((ROWS * 32 + 255) / 256), 256, 0, stream>>>(q, k, qn, out);
    norm_qb_kernel<<<dim3(ROWS), 256, 0, stream>>>(qbb, qb);
    gemm_qneg_kernel<<<dim3(NCB1, ROWS / 128), 256, 0, stream>>>(qn, queue, out, smm, sms);
    gemm_qb_topk_kernel<<<dim3(G2_NSB, ROWS / 64), 256, 0, stream>>>(qb, queue_b, topv, topi);
    topk_merge_kernel<<<dim3((ROWS + 255) / 256), 256, 0, stream>>>(topv, topi, knn);
    row_lse_loss_kernel<<<dim3((ROWS + 255) / 256), 256, 0, stream>>>(smm, sms, knn, out, row_loss);
    finalize_kernel<<<dim3(1), 256, 0, stream>>>(row_loss, out);
}
